// Encoder_51247549775991
// MI455X (gfx1250) — matrix operand order repaired and checked by reading, not compiled, not run on hardware
//
#include <hip/hip_runtime.h>

typedef __attribute__((ext_vector_type(16))) _Float16 v16h;
typedef __attribute__((ext_vector_type(8)))  _Float16 v8h;
typedef __attribute__((ext_vector_type(4)))  _Float16 v4h;
typedef __attribute__((ext_vector_type(8)))  float    v8f;

#define NPTS  16384
#define MANC  4096
#define PLOC  4
#define PSP   16
#define NCLOC (MANC*PLOC)   /* 16384 */
#define NCSP  (MANC*PSP)    /* 65536 */
#define FEATN 4096
#define FINF  3.4028235e38f

// ---------------------------------------------------------------------------
// pts[i][c] = xyz[c][i]
// ---------------------------------------------------------------------------
__global__ void k_make_pts(const float* __restrict__ xyz, float* __restrict__ pts) {
  int i = blockIdx.x * blockDim.x + threadIdx.x;
  if (i < NPTS) {
    pts[i * 3 + 0] = xyz[i];
    pts[i * 3 + 1] = xyz[NPTS + i];
    pts[i * 3 + 2] = xyz[2 * NPTS + i];
  }
}

// ---------------------------------------------------------------------------
// Furthest point sampling: single block, dist[] lives in LDS (64 KB of the
// 320 KB WGP pool), cross-wave argmax through a small global scratch.
// Writes sparse[M][3] and the transposed first output d_out[3][M].
// ---------------------------------------------------------------------------
__global__ __launch_bounds__(1024)
void k_fps(const float* __restrict__ pts, float* __restrict__ sparse,
           float* __restrict__ outT, float* __restrict__ red) {
  __shared__ float dist[NPTS];
  const int tid  = threadIdx.x;
  const int lane = tid & 31;
  const int wv   = tid >> 5;
  float*          redV = red;              // 32 floats
  int*            redI = (int*)(red + 32); // 32 ints
  volatile float* bc   = red + 72;         // 3 floats broadcast

  for (int i = tid; i < NPTS; i += 1024) dist[i] = FINF;
  float cx = pts[0], cy = pts[1], cz = pts[2];
  if (tid == 0) {
    sparse[0] = cx; sparse[1] = cy; sparse[2] = cz;
    outT[0] = cx; outT[MANC] = cy; outT[2 * MANC] = cz;
  }
  __syncthreads();

  for (int it = 1; it < MANC; ++it) {
    float bv = -1.0f; int bi = 0x7fffffff;
    for (int i = tid; i < NPTS; i += 1024) {
      float dx = pts[i * 3 + 0] - cx, dy = pts[i * 3 + 1] - cy, dz = pts[i * 3 + 2] - cz;
      float d = fminf(dist[i], dx * dx + dy * dy + dz * dz);
      dist[i] = d;
      if (d > bv) { bv = d; bi = i; }
    }
#pragma unroll
    for (int off = 16; off > 0; off >>= 1) {
      float ov = __shfl_down(bv, off, 32);
      int   oi = __shfl_down(bi, off, 32);
      if (ov > bv || (ov == bv && oi < bi)) { bv = ov; bi = oi; }
    }
    if (lane == 0) { redV[wv] = bv; redI[wv] = bi; }
    __threadfence();
    __syncthreads();
    if (tid < 32) {
      float v = redV[tid]; int ii = redI[tid];
#pragma unroll
      for (int off = 16; off > 0; off >>= 1) {
        float ov = __shfl_down(v, off, 32);
        int   oi = __shfl_down(ii, off, 32);
        if (ov > v || (ov == v && oi < ii)) { v = ov; ii = oi; }
      }
      if (tid == 0) {
        float x = pts[ii * 3], y = pts[ii * 3 + 1], z = pts[ii * 3 + 2];
        sparse[it * 3] = x; sparse[it * 3 + 1] = y; sparse[it * 3 + 2] = z;
        outT[it] = x; outT[MANC + it] = y; outT[2 * MANC + it] = z;
        bc[0] = x; bc[1] = y; bc[2] = z;
      }
    }
    __threadfence();
    __syncthreads();
    cx = bc[0]; cy = bc[1]; cz = bc[2];
  }
}

// ---------------------------------------------------------------------------
// KNN of anchors against the full cloud; drop self; write centered coords as
// H0 [3][NCLOC] column = m*4 + j.
// ---------------------------------------------------------------------------
__global__ void k_knn_local(const float* __restrict__ pts, const float* __restrict__ sparse,
                            float* __restrict__ H0) {
  int m = blockIdx.x * blockDim.x + threadIdx.x;
  if (m >= MANC) return;
  float ax = sparse[m * 3], ay = sparse[m * 3 + 1], az = sparse[m * 3 + 2];
  float bd[5]; int bi[5];
#pragma unroll
  for (int q = 0; q < 5; ++q) { bd[q] = FINF; bi[q] = 0; }
  for (int i = 0; i < NPTS; ++i) {
    float dx = pts[i * 3] - ax, dy = pts[i * 3 + 1] - ay, dz = pts[i * 3 + 2] - az;
    float d = dx * dx + dy * dy + dz * dz;
    if (d < bd[4]) {
      bd[4] = d; bi[4] = i;
#pragma unroll
      for (int q = 4; q > 0; --q)
        if (bd[q] < bd[q - 1]) {
          float tv = bd[q]; bd[q] = bd[q - 1]; bd[q - 1] = tv;
          int   ti = bi[q]; bi[q] = bi[q - 1]; bi[q - 1] = ti;
        }
    }
  }
#pragma unroll
  for (int j = 1; j <= 4; ++j) {
    int col = m * PLOC + (j - 1); int ii = bi[j];
    H0[0 * NCLOC + col] = pts[ii * 3 + 0] - ax;
    H0[1 * NCLOC + col] = pts[ii * 3 + 1] - ay;
    H0[2 * NCLOC + col] = pts[ii * 3 + 2] - az;
  }
}

// Same among the sparse anchors, top-17 drop self -> H0 [3][NCSP]
__global__ void k_knn_anchor(const float* __restrict__ sparse, float* __restrict__ H0) {
  int m = blockIdx.x * blockDim.x + threadIdx.x;
  if (m >= MANC) return;
  float ax = sparse[m * 3], ay = sparse[m * 3 + 1], az = sparse[m * 3 + 2];
  float bd[17]; int bi[17];
#pragma unroll
  for (int q = 0; q < 17; ++q) { bd[q] = FINF; bi[q] = 0; }
  for (int i = 0; i < MANC; ++i) {
    float dx = sparse[i * 3] - ax, dy = sparse[i * 3 + 1] - ay, dz = sparse[i * 3 + 2] - az;
    float d = dx * dx + dy * dy + dz * dz;
    if (d < bd[16]) {
      bd[16] = d; bi[16] = i;
#pragma unroll
      for (int q = 16; q > 0; --q)
        if (bd[q] < bd[q - 1]) {
          float tv = bd[q]; bd[q] = bd[q - 1]; bd[q - 1] = tv;
          int   ti = bi[q]; bi[q] = bi[q - 1]; bi[q - 1] = ti;
        }
    }
  }
#pragma unroll
  for (int j = 1; j <= 16; ++j) {
    int col = m * PSP + (j - 1); int ii = bi[j];
    H0[0 * (size_t)NCSP + col] = sparse[ii * 3 + 0] - ax;
    H0[1 * (size_t)NCSP + col] = sparse[ii * 3 + 1] - ay;
    H0[2 * (size_t)NCSP + col] = sparse[ii * 3 + 2] - az;
  }
}

// ---------------------------------------------------------------------------
// Per-patch feature-space top-4 KNN (sparse path only, patches of 16 points).
// ---------------------------------------------------------------------------
__global__ void k_feat_knn(const float* __restrict__ H, int D, int* __restrict__ idx) {
  int t = blockIdx.x * blockDim.x + threadIdx.x;
  if (t >= MANC * PSP) return;
  int p = t & 15, base = t & ~15;
  float acc[16];
#pragma unroll
  for (int j = 0; j < 16; ++j) acc[j] = 0.f;
  for (int c = 0; c < D; ++c) {
    const float* row = H + (size_t)c * NCSP + base;
    float hp = row[p];
#pragma unroll
    for (int j = 0; j < 16; ++j) { float df = hp - row[j]; acc[j] += df * df; }
  }
  float bd[4]; int bi[4];
#pragma unroll
  for (int q = 0; q < 4; ++q) { bd[q] = FINF; bi[q] = 0; }
#pragma unroll
  for (int j = 0; j < 16; ++j) {
    float d = acc[j];
    if (d < bd[3]) {
      bd[3] = d; bi[3] = j;
#pragma unroll
      for (int q = 3; q > 0; --q)
        if (bd[q] < bd[q - 1]) {
          float tv = bd[q]; bd[q] = bd[q - 1]; bd[q - 1] = tv;
          int   ti = bi[q]; bi[q] = bi[q - 1]; bi[q - 1] = ti;
        }
    }
  }
#pragma unroll
  for (int q = 0; q < 4; ++q) idx[t * 4 + q] = bi[q];
}

// ---------------------------------------------------------------------------
// Edge-conv epilogue: H[o][col] = relu(max_j (A[o][nbr_j] + C[o][col]))
// AC rows 0..O-1 = A (neighbor half), rows O..2O-1 = C (center half).
// ---------------------------------------------------------------------------
__global__ void k_edge_max_loc(const float* __restrict__ AC, float* __restrict__ Hout,
                               int O, int Nc) {
  size_t t = (size_t)blockIdx.x * blockDim.x + threadIdx.x;
  if (t >= (size_t)O * Nc) return;
  int col = (int)(t % Nc); int o = (int)(t / Nc);
  const float* A = AC + (size_t)o * Nc;
  float cv = A[(size_t)O * Nc + col];
  int mb = col & ~(PLOC - 1);      // k == P: all points of the patch are neighbors
  float v = A[mb] + cv;
#pragma unroll
  for (int j = 1; j < PLOC; ++j) v = fmaxf(v, A[mb + j] + cv);
  Hout[(size_t)o * Nc + col] = fmaxf(v, 0.f);
}

__global__ void k_edge_max_sp(const float* __restrict__ AC, const int* __restrict__ idx,
                              float* __restrict__ Hout, int O, int Nc) {
  size_t t = (size_t)blockIdx.x * blockDim.x + threadIdx.x;
  if (t >= (size_t)O * Nc) return;
  int col = (int)(t % Nc); int o = (int)(t / Nc);
  const float* A = AC + (size_t)o * Nc;
  float cv = A[(size_t)O * Nc + col];
  int mb = col & ~(PSP - 1);
  const int* id = idx + (size_t)col * 4;
  float v = -FINF;
#pragma unroll
  for (int j = 0; j < 4; ++j) v = fmaxf(v, A[mb + id[j]] + cv);
  Hout[(size_t)o * Nc + col] = fmaxf(v, 0.f);
}

// tanh(max_p y) per (channel, patch): tanh is monotone so tanh after max.
__global__ void k_tanh_max(const float* __restrict__ Y, float* __restrict__ feat,
                           int P, int Nc) {
  int t = blockIdx.x * blockDim.x + threadIdx.x;
  if (t >= 512 * MANC) return;
  int m = t % MANC, o = t / MANC;
  const float* y = Y + (size_t)o * Nc + (size_t)m * P;
  float v = y[0];
  for (int p = 1; p < P; ++p) v = fmaxf(v, y[p]);
  feat[(size_t)o * FEATN + m] = tanhf(v);
}

// ---------------------------------------------------------------------------
// WMMA GEMM:  Y[Mrows][Ncols] = W[Mrows][Kdim] @ X[Kdim][Ncols]   (f32 I/O,
// f16 compute, f32 accumulate).  KDIM is a compile-time constant so all tile
// bounds fold away (only KDIM=3 has a tail, handled branchlessly via clamped
// addresses + selects).  128 threads = 4 waves; block tile = 64 rows x 32
// cols.  Both operand slabs are staged cooperatively into LDS with coalesced
// float4 (b128) loads + f32->f16 convert; fragments come from contiguous
// ds reads in k = 16*half + e order, the same for A and B.  Each wave
// holds one A fragment and two B fragments -> 2 WMMAs per K-step.
//   EPI: 0 = none, 1 = +bias, 2 = +bias, relu
// Mrows multiple of 16; Ncols multiple of 32.
// ---------------------------------------------------------------------------
template <int KDIM, int EPI>
__global__ __launch_bounds__(128)
void gemm_wmma(const float* __restrict__ W, int ldw,
               const float* __restrict__ X, float* __restrict__ Y,
               const float* __restrict__ bias, int Mrows, int Ncols) {
  constexpr int  KT   = (KDIM + 31) / 32;
  constexpr bool TAIL = (KDIM % 32) != 0;
  __shared__ __align__(64) _Float16 xs[32 * 32];   // [n][k] f16
  __shared__ __align__(64) _Float16 wsA[64 * 32];  // [r][k] f16
  const int tid     = threadIdx.x;
  const int wave    = tid >> 5;
  const int lane    = tid & 31;
  const int colBase = blockIdx.x * 32;
  const int rowBase = blockIdx.y * 64;
  const int tileM   = rowBase + wave * 16;
  const bool active = tileM < Mrows;
  const int halfSel = (lane < 16) ? 0 : 1;
  const int a0      = halfSel * 8;
  const int ln      = lane & 15;

  v8f acc0 = {}, acc1 = {};

#pragma unroll
  for (int kt = 0; kt < KT; ++kt) {
    const int kBase = kt * 32;
    if constexpr (!TAIL) {
      // X slab: 32 K x 32 N = 256 float4, 2 per thread, coalesced b128
#pragma unroll
      for (int q = 0; q < 2; ++q) {
        int i = tid + q * 128;
        int k = i >> 3, n4 = (i & 7) * 4;
        const float4 f = *(const float4*)&X[(size_t)(kBase + k) * Ncols + colBase + n4];
        xs[(n4 + 0) * 32 + k] = (_Float16)f.x;
        xs[(n4 + 1) * 32 + k] = (_Float16)f.y;
        xs[(n4 + 2) * 32 + k] = (_Float16)f.z;
        xs[(n4 + 3) * 32 + k] = (_Float16)f.w;
      }
      // W slab: 64 rows x 32 K = 512 float4, 4 per thread
#pragma unroll
      for (int q = 0; q < 4; ++q) {
        int i = tid + q * 128;
        int r = i >> 3, s4 = (i & 7) * 4;
        int rr = rowBase + r; if (rr >= Mrows) rr = Mrows - 1;  // clamp (unused rows)
        const float4 f = *(const float4*)&W[(size_t)rr * ldw + kBase + s4];
        v4h h; h[0] = (_Float16)f.x; h[1] = (_Float16)f.y;
               h[2] = (_Float16)f.z; h[3] = (_Float16)f.w;
        *(v4h*)&wsA[r * 32 + s4] = h;
      }
    } else {
      // branchless scalar staging (KDIM=3 only): clamped address + select
#pragma unroll
      for (int q = 0; q < 8; ++q) {
        int i = tid + q * 128;           // 1024 X elements
        int n = i & 31, k = i >> 5;
        int kc = kBase + k;
        int kl = (kc < KDIM) ? kc : (KDIM - 1);
        float v = X[(size_t)kl * Ncols + colBase + n];
        xs[n * 32 + k] = (_Float16)((kc < KDIM) ? v : 0.f);
      }
#pragma unroll
      for (int q = 0; q < 16; ++q) {
        int i = tid + q * 128;           // 2048 W elements
        int r = i >> 5, k = i & 31;
        int rr = rowBase + r; if (rr >= Mrows) rr = Mrows - 1;
        int kc = kBase + k;
        int kl = (kc < KDIM) ? kc : (KDIM - 1);
        float v = W[(size_t)rr * ldw + kl];
        wsA[r * 32 + k] = (_Float16)((kc < KDIM) ? v : 0.f);
      }
    }
    if (kt + 1 < KT)                     // pull next X slab toward the WGP
      __builtin_prefetch(&X[(size_t)(kBase + 32) * Ncols + colBase], 0, 1);
    __syncthreads();
    {
      // A fragment: row = ln; elem e <-> K = 16*halfSel + e
      const int row32 = (wave * 16 + ln) * 32;
      v16h afrag = *(const v16h*)&wsA[row32 + halfSel * 16];
      // B fragments: col = ln (+16); elem e <-> K = 16*halfSel + e
      v16h bf0 = *(const v16h*)&xs[ln * 32 + halfSel * 16];
      v16h bf1 = *(const v16h*)&xs[(ln + 16) * 32 + halfSel * 16];
      acc0 = __builtin_amdgcn_wmma_f32_16x16x32_f16(false, afrag, false, bf0,
                                                    (short)0, acc0, false, false);
      acc1 = __builtin_amdgcn_wmma_f32_16x16x32_f16(false, afrag, false, bf1,
                                                    (short)0, acc1, false, false);
    }
    __syncthreads();
  }

  if (active) {
    const int col0 = colBase + ln;
#pragma unroll
    for (int v = 0; v < 8; ++v) {
      int row = tileM + v + halfSel * 8;   // C/D layout: VGPR v -> M v|v+8
      float v0 = acc0[v], v1 = acc1[v];
      if (EPI >= 1) { float b = bias[row]; v0 += b; v1 += b; }
      if (EPI == 2) { v0 = fmaxf(v0, 0.f); v1 = fmaxf(v1, 0.f); }
      Y[(size_t)row * Ncols + col0]      = v0;
      Y[(size_t)row * Ncols + col0 + 16] = v1;
    }
  }
}

// ---------------------------------------------------------------------------
// host side
// ---------------------------------------------------------------------------
template <int KDIM, int EPI>
static void launch_gemm(hipStream_t s, const float* W, int ldw,
                        const float* X, float* Y, const float* b, int Mr, int Nc) {
  dim3 g(Nc / 32, (Mr + 63) / 64);
  gemm_wmma<KDIM, EPI><<<g, dim3(128), 0, s>>>(W, ldw, X, Y, b, Mr, Nc);
}

static inline unsigned blks(size_t n, int b) { return (unsigned)((n + b - 1) / b); }

extern "C" void kernel_launch(void* const* d_in, const int* in_sizes, int n_in,
                              void* d_out, int out_size, void* d_ws, size_t ws_size,
                              hipStream_t stream) {
  (void)in_sizes; (void)n_in; (void)out_size; (void)ws_size;
  const float* lw[5]; const float* sw[5];
  const float* xyz = (const float*)d_in[0];
  for (int j = 0; j < 5; ++j) { lw[j] = (const float*)d_in[1 + j]; sw[j] = (const float*)d_in[6 + j]; }
  const float* wf1 = (const float*)d_in[11];
  const float* bf1 = (const float*)d_in[12];
  const float* wf2 = (const float*)d_in[13];
  const float* bf2 = (const float*)d_in[14];
  const float* wo  = (const float*)d_in[15];
  const float* bo  = (const float*)d_in[16];
  float* out = (float*)d_out;

  float* ws     = (float*)d_ws;
  float* pts    = ws;  ws += 3 * NPTS;
  float* sparse = ws;  ws += 3 * MANC;
  float* red    = ws;  ws += 128;
  float* H0     = ws;  ws += (size_t)3 * NCSP;
  float* Hcat   = ws;  ws += (size_t)512 * NCSP;
  float* AC     = ws;  ws += (size_t)512 * NCSP;
  int*   knn    = (int*)ws; ws += (size_t)MANC * PSP * 4;
  float* feat   = ws;  ws += (size_t)1024 * FEATN;
  float* h1     = ws;  ws += (size_t)512 * FEATN;
  float* h2     = ws;  ws += (size_t)256 * FEATN;

  k_make_pts<<<NPTS / 256, 256, 0, stream>>>(xyz, pts);
  k_fps<<<1, 1024, 0, stream>>>(pts, sparse, out, red);

  // ---------------- local path (P=4, Nc=16384) ----------------
  {
    const int Nc = NCLOC;
    k_knn_local<<<MANC / 128, 128, 0, stream>>>(pts, sparse, H0);
    // L1 (O=64, D=3)
    launch_gemm<3, 0>(stream, lw[0],     6, H0, AC,                   nullptr, 64, Nc);
    launch_gemm<3, 0>(stream, lw[0] + 3, 6, H0, AC + (size_t)64 * Nc, nullptr, 64, Nc);
    k_edge_max_loc<<<blks((size_t)64 * Nc, 256), 256, 0, stream>>>(AC, Hcat, 64, Nc);
    // L2 (O=64, D=64)
    launch_gemm<64, 0>(stream, lw[1],      128, Hcat, AC,                   nullptr, 64, Nc);
    launch_gemm<64, 0>(stream, lw[1] + 64, 128, Hcat, AC + (size_t)64 * Nc, nullptr, 64, Nc);
    k_edge_max_loc<<<blks((size_t)64 * Nc, 256), 256, 0, stream>>>(AC, Hcat + (size_t)64 * Nc, 64, Nc);
    // L3 (O=128, D=64)
    launch_gemm<64, 0>(stream, lw[2],      128, Hcat + (size_t)64 * Nc, AC,                    nullptr, 128, Nc);
    launch_gemm<64, 0>(stream, lw[2] + 64, 128, Hcat + (size_t)64 * Nc, AC + (size_t)128 * Nc, nullptr, 128, Nc);
    k_edge_max_loc<<<blks((size_t)128 * Nc, 256), 256, 0, stream>>>(AC, Hcat + (size_t)128 * Nc, 128, Nc);
    // L4 (O=256, D=128)
    launch_gemm<128, 0>(stream, lw[3],       256, Hcat + (size_t)128 * Nc, AC,                    nullptr, 256, Nc);
    launch_gemm<128, 0>(stream, lw[3] + 128, 256, Hcat + (size_t)128 * Nc, AC + (size_t)256 * Nc, nullptr, 256, Nc);
    k_edge_max_loc<<<blks((size_t)256 * Nc, 256), 256, 0, stream>>>(AC, Hcat + (size_t)256 * Nc, 256, Nc);
    // L5 (512x512 on the concatenated 512 channels) + tanh + max over P
    launch_gemm<512, 0>(stream, lw[4], 512, Hcat, AC, nullptr, 512, Nc);
    k_tanh_max<<<blks((size_t)512 * MANC, 256), 256, 0, stream>>>(AC, feat + (size_t)512 * FEATN, PLOC, Nc);
  }

  // ---------------- sparse/anchor path (P=16, Nc=65536) ----------------
  {
    const int Nc = NCSP;
    k_knn_anchor<<<MANC / 128, 128, 0, stream>>>(sparse, H0);
    // L1
    k_feat_knn<<<blks((size_t)MANC * PSP, 256), 256, 0, stream>>>(H0, 3, knn);
    launch_gemm<3, 0>(stream, sw[0],     6, H0, AC,                   nullptr, 64, Nc);
    launch_gemm<3, 0>(stream, sw[0] + 3, 6, H0, AC + (size_t)64 * Nc, nullptr, 64, Nc);
    k_edge_max_sp<<<blks((size_t)64 * Nc, 256), 256, 0, stream>>>(AC, knn, Hcat, 64, Nc);
    // L2
    k_feat_knn<<<blks((size_t)MANC * PSP, 256), 256, 0, stream>>>(Hcat, 64, knn);
    launch_gemm<64, 0>(stream, sw[1],      128, Hcat, AC,                   nullptr, 64, Nc);
    launch_gemm<64, 0>(stream, sw[1] + 64, 128, Hcat, AC + (size_t)64 * Nc, nullptr, 64, Nc);
    k_edge_max_sp<<<blks((size_t)64 * Nc, 256), 256, 0, stream>>>(AC, knn, Hcat + (size_t)64 * Nc, 64, Nc);
    // L3
    k_feat_knn<<<blks((size_t)MANC * PSP, 256), 256, 0, stream>>>(Hcat + (size_t)64 * Nc, 64, knn);
    launch_gemm<64, 0>(stream, sw[2],      128, Hcat + (size_t)64 * Nc, AC,                    nullptr, 128, Nc);
    launch_gemm<64, 0>(stream, sw[2] + 64, 128, Hcat + (size_t)64 * Nc, AC + (size_t)128 * Nc, nullptr, 128, Nc);
    k_edge_max_sp<<<blks((size_t)128 * Nc, 256), 256, 0, stream>>>(AC, knn, Hcat + (size_t)128 * Nc, 128, Nc);
    // L4
    k_feat_knn<<<blks((size_t)MANC * PSP, 256), 256, 0, stream>>>(Hcat + (size_t)128 * Nc, 128, knn);
    launch_gemm<128, 0>(stream, sw[3],       256, Hcat + (size_t)128 * Nc, AC,                    nullptr, 256, Nc);
    launch_gemm<128, 0>(stream, sw[3] + 128, 256, Hcat + (size_t)128 * Nc, AC + (size_t)256 * Nc, nullptr, 256, Nc);
    k_edge_max_sp<<<blks((size_t)256 * Nc, 256), 256, 0, stream>>>(AC, knn, Hcat + (size_t)256 * Nc, 256, Nc);
    // L5 + tanh + max
    launch_gemm<512, 0>(stream, sw[4], 512, Hcat, AC, nullptr, 512, Nc);
    k_tanh_max<<<blks((size_t)512 * MANC, 256), 256, 0, stream>>>(AC, feat, PSP, Nc);
  }

  // ---------------- MLP head on feat [1024][4096] ----------------
  launch_gemm<1024, 2>(stream, wf1, 1024, feat, h1, bf1, 512, FEATN);
  launch_gemm< 512, 2>(stream, wf2,  512, h1,   h2, bf2, 256, FEATN);
  launch_gemm< 256, 1>(stream, wo,   256, h2, out + (size_t)3 * MANC, bo, 16, FEATN);
}
